// LSTMModel3_90623809945618
// MI455X (gfx1250) — matrix operand order repaired and checked by reading, not compiled, not run on hardware
//
#include <hip/hip_runtime.h>
#include <math.h>

// ---------------- CDNA5 WMMA types ----------------
typedef __bf16 bf16_t;
typedef __attribute__((ext_vector_type(16))) __bf16 v16bf;
typedef __attribute__((ext_vector_type(4)))  __bf16 v4bf;
typedef __attribute__((ext_vector_type(8)))  float  v8f;

#define HDIM   256
#define FOURH  1024
#define BATCH  512
#define TSTEPS 512
#define DIN    128
#define MT     32            // batch rows per block (2 x 16-row fragments)

__device__ __forceinline__ v8f wmma_bf16(v16bf a, v16bf b, v8f c) {
    return __builtin_amdgcn_wmma_f32_16x16x32_bf16(false, a, false, b,
                                                   (short)0, c, false, false);
}

// ===========================================================================
// Fragment layouts (bf16, 32B per lane per tile, perfectly coalesced):
//  A-fragment: value[lane][j] = A[r16*16 + (lane&15)][kt*32 + kmap(j, lane>>4)]
//              kmap(j,hi) = j + 8*hi + 8*(j>=8)
//              flat: ((r16*KT + kt)*32 + lane)*16 + j
//  B-fragment: value[lane][j] = W[kt*32 + kmap(j, lane>>4)][c16*16 + (lane&15)]
//              flat: ((c16*KT + kt)*32 + lane)*16 + j
// ===========================================================================

// Pack one f32 row-major matrix [rows x 32*KT] (leading dim lda) into
// A-fragment order. One thread per (r16, kt, lane) -> 16 bf16 (32B store).
__global__ void __launch_bounds__(256)
pack_a(const float* __restrict__ src, long lda, int KT, bf16_t* __restrict__ dst)
{
    const int t    = blockIdx.x * blockDim.x + threadIdx.x;
    const int lane = t & 31;
    const int kt   = (t >> 5) % KT;
    const int r16  = t / (32 * KT);
    const int m = lane & 15, hi = lane >> 4;
    const float* s = src + (long)(r16 * 16 + m) * lda + kt * 32;
    v16bf v;
    #pragma unroll
    for (int j = 0; j < 16; ++j) {
        int kk = j + hi * 8 + ((j >= 8) ? 8 : 0);
        v[j] = (bf16_t)s[kk];
    }
    *(v16bf*)(dst + (long)t * 16) = v;
}

// Pack x [B, T, D] into per-step A-fragment order:
//   dst[((t*32 + r16)*4 + kt)*512 + lane*16 + j]
__global__ void __launch_bounds__(256)
pack_x(const float* __restrict__ x, bf16_t* __restrict__ dst)
{
    const long t   = (long)blockIdx.x * blockDim.x + threadIdx.x;
    const int lane = (int)(t & 31);
    const int kt   = (int)((t >> 5) & 3);
    const int r16  = (int)((t >> 7) & 31);
    const int ts   = (int)(t >> 12);
    const int m = lane & 15, hi = lane >> 4;
    const float* s = x + ((long)(r16 * 16 + m) * TSTEPS + ts) * DIN + kt * 32;
    v16bf v;
    #pragma unroll
    for (int j = 0; j < 16; ++j) {
        int kk = j + hi * 8 + ((j >= 8) ? 8 : 0);
        v[j] = (bf16_t)s[kk];
    }
    *(v16bf*)(dst + t * 16) = v;
}

// Pack weights [32*KT x 4H] into B-fragment order.
__global__ void __launch_bounds__(256)
pack_w(const float* __restrict__ W, int KT, bf16_t* __restrict__ dst)
{
    const int t    = blockIdx.x * blockDim.x + threadIdx.x;
    const int lane = t & 31;
    const int kt   = (t >> 5) % KT;
    const int c16  = t / (32 * KT);
    const int lo = lane & 15, hi = lane >> 4;
    const float* s = W + (long)(kt * 32 + 8 * hi) * FOURH + c16 * 16 + lo;
    v16bf v;
    #pragma unroll
    for (int j = 0; j < 16; ++j)
        v[j] = (bf16_t)s[(long)(j + ((j >= 8) ? 8 : 0)) * FOURH];
    *(v16bf*)(dst + (long)t * 16) = v;
}

// ---------------------------------------------------------------------------
// acc{0,1} += A[32 x 32*KT] * W[32*KT x 16], everything pre-packed bf16.
// Per K-tile: 6x global_load_b128, 2x v_wmma. No LDS, no barriers.
// ---------------------------------------------------------------------------
template<int KT>
__device__ __forceinline__ void gemm_pass_frag(
    const bf16_t* __restrict__ Ap, const bf16_t* __restrict__ Wp,
    int r16base, int cblk, int lane, v8f& acc0, v8f& acc1)
{
    #pragma unroll
    for (int i = 0; i < KT; ++i) {
        v16bf a0 = *(const v16bf*)(Ap + ((long)((r16base + 0) * KT + i) * 32 + lane) * 16);
        v16bf a1 = *(const v16bf*)(Ap + ((long)((r16base + 1) * KT + i) * 32 + lane) * 16);
        v16bf b  = *(const v16bf*)(Wp + ((long)(cblk * KT + i) * 32 + lane) * 16);
        acc0 = wmma_bf16(a0, b, acc0);
        acc1 = wmma_bf16(a1, b, acc1);
    }
}

// ---------------------------------------------------------------------------
// One fused LSTM cell step:
//   gates = in1 @ W1 + h_prev @ W2 + bias ; i,f,o = sigmoid ; g = tanh
//   c = f*c + i*g ; h_out = o*tanh(c)   (h_out written in A-fragment bf16)
// Grid: (BATCH/32, HDIM/16); block: 128 threads = 4 waves (one per gate).
// ---------------------------------------------------------------------------
template<int K1>
__global__ void __launch_bounds__(128)
lstm_step(const bf16_t* __restrict__ in1p,
          const bf16_t* __restrict__ Wp1,
          const bf16_t* __restrict__ in2p,
          const bf16_t* __restrict__ Wp2,
          const float*  __restrict__ bias,
          float*  __restrict__ c_state,
          bf16_t* __restrict__ h_out)
{
    __shared__ float lds_g[4 * MT * 16];       // gate exchange, 8 KB

    const int tid  = threadIdx.x;
    const int w    = tid >> 5;                 // wave id == gate (i,f,g,o)
    const int lane = tid & 31;
    const int lo   = lane & 15;
    const int m0   = blockIdx.x * MT;
    const int n0   = blockIdx.y * 16;
    const int cblk = w * 16 + blockIdx.y;      // 16-col block inside 4H

    const float bv = bias[cblk * 16 + lo];     // C/D element (r,lane): N = lo
    v8f acc0 = { bv, bv, bv, bv, bv, bv, bv, bv };
    v8f acc1 = acc0;

    gemm_pass_frag<K1 / 32>(in1p, Wp1, blockIdx.x * 2, cblk, lane, acc0, acc1);
    gemm_pass_frag<HDIM / 32>(in2p, Wp2, blockIdx.x * 2, cblk, lane, acc0, acc1);

    // Exchange gates through LDS (only barrier in the kernel).
    const int hi = lane >> 4;
    #pragma unroll
    for (int r = 0; r < 8; ++r) {              // element (M = r+8*hi, N = lo)
        lds_g[w * (MT * 16) + (r + 8 * hi) * 16 + lo]      = acc0[r];
        lds_g[w * (MT * 16) + (16 + r + 8 * hi) * 16 + lo] = acc1[r];
    }
    __syncthreads();

    // Fused cell update: thread -> (m = tid>>2, 4 consecutive n).
    {
        const int m  = tid >> 2;               // 0..31
        const int n4 = (tid & 3) * 4;          // 0,4,8,12
        const int e  = m * 16 + n4;
        float4 gi = *(const float4*)(lds_g + 0 * (MT * 16) + e);
        float4 gf = *(const float4*)(lds_g + 1 * (MT * 16) + e);
        float4 gg = *(const float4*)(lds_g + 2 * (MT * 16) + e);
        float4 go = *(const float4*)(lds_g + 3 * (MT * 16) + e);

        const long cidx = (long)(m0 + m) * HDIM + n0 + n4;
        float4 cold = *(const float4*)(c_state + cidx);

        float cv[4], hv[4];
        float gI[4] = { gi.x, gi.y, gi.z, gi.w };
        float gF[4] = { gf.x, gf.y, gf.z, gf.w };
        float gG[4] = { gg.x, gg.y, gg.z, gg.w };
        float gO[4] = { go.x, go.y, go.z, go.w };
        float cO[4] = { cold.x, cold.y, cold.z, cold.w };
        #pragma unroll
        for (int u = 0; u < 4; ++u) {
            float it = 1.0f / (1.0f + __expf(-gI[u]));
            float ft = 1.0f / (1.0f + __expf(-gF[u]));
            float gt = tanhf(gG[u]);
            float ot = 1.0f / (1.0f + __expf(-gO[u]));
            cv[u] = ft * cO[u] + it * gt;
            hv[u] = ot * tanhf(cv[u]);
        }
        *(float4*)(c_state + cidx) = make_float4(cv[0], cv[1], cv[2], cv[3]);

        // h -> A-fragment layout (KT = 8 for H=256):
        //   r16 = M/16, kt = n0>>5, lane' = (M&15) + 16*(n>=8),
        //   j = (n&7) + 8*((n0>>4)&1); n4-group has uniform hi, consecutive j.
        const int M    = m0 + m;
        const int r16  = M >> 4;
        const int kt   = n0 >> 5;
        const int lnp  = (M & 15) + 16 * (n4 >= 8 ? 1 : 0);
        const int j0   = (n4 & 7) + 8 * ((n0 >> 4) & 1);
        v4bf hp;
        hp[0] = (bf16_t)hv[0]; hp[1] = (bf16_t)hv[1];
        hp[2] = (bf16_t)hv[2]; hp[3] = (bf16_t)hv[3];
        *(v4bf*)(h_out + ((long)(r16 * 8 + kt) * 32 + lnp) * 16 + j0) = hp;
    }
}

// ---------------------------------------------------------------------------
// Final FC head: out[b] = h[b,:] . fc_w[:,0] + fc_b, h in A-fragment bf16.
// ---------------------------------------------------------------------------
__global__ void __launch_bounds__(256)
fc_head(const bf16_t* __restrict__ hp, const float* __restrict__ fcw,
        const float* __restrict__ fcb, float* __restrict__ out)
{
    __shared__ float red[256];
    const int b = blockIdx.x, k = threadIdx.x;
    const int r16 = b >> 4, kt = k >> 5;
    const int lnp = (b & 15) + 16 * (((k & 15) >= 8) ? 1 : 0);
    const int j   = (k & 7) + 8 * (((k & 31) >= 16) ? 1 : 0);
    red[k] = (float)hp[((long)(r16 * 8 + kt) * 32 + lnp) * 16 + j] * fcw[k];
    __syncthreads();
    for (int s = 128; s > 0; s >>= 1) {
        if (k < s) red[k] += red[k + s];
        __syncthreads();
    }
    if (k == 0) out[b] = red[0] + fcb[0];
}

// ---------------------------------------------------------------------------
extern "C" void kernel_launch(void* const* d_in, const int* in_sizes, int n_in,
                              void* d_out, int out_size, void* d_ws, size_t ws_size,
                              hipStream_t stream) {
    const float* x    = (const float*)d_in[0];   // [B, T, D]
    const float* h0in = (const float*)d_in[1];   // [L, B, H]
    const float* c0in = (const float*)d_in[2];   // [L, B, H]
    const float* wi0  = (const float*)d_in[3];   // [D, 4H]
    const float* hi0  = (const float*)d_in[4];   // [H, 4H]
    const float* b0   = (const float*)d_in[5];   // [4H]
    const float* wi1  = (const float*)d_in[6];   // [H, 4H]
    const float* hi1  = (const float*)d_in[7];   // [H, 4H]
    const float* b1   = (const float*)d_in[8];   // [4H]
    const float* fcw  = (const float*)d_in[9];   // [H, 1]
    const float* fcb  = (const float*)d_in[10];  // [1]

    const long BH = (long)BATCH * HDIM;          // 131072
    float* ws  = (float*)d_ws;
    float* c0w = ws;                             // f32 cell state, layer 0
    float* c1w = ws + BH;                        // f32 cell state, layer 1

    bf16_t* bb = (bf16_t*)(ws + 2 * BH);
    bf16_t* h0b[2] = { bb,          bb + BH };   // h layer0, A-frag bf16
    bf16_t* h1b[2] = { bb + 2 * BH, bb + 3 * BH };
    bf16_t* wp0 = bb + 4 * BH;                   // [D  x4H] B-frag: 131072
    bf16_t* wh0 = wp0 + (long)DIN * FOURH;       // [H x 4H] B-frag: 262144
    bf16_t* wp1 = wh0 + (long)HDIM * FOURH;
    bf16_t* wh1 = wp1 + (long)HDIM * FOURH;
    bf16_t* xp  = wh1 + (long)HDIM * FOURH;      // packed x: 33.5M bf16

    // ---- One-time (per call) packing: weights, x, initial h ----
    pack_w<<<dim3((64 * (DIN / 32) * 32) / 256),  dim3(256), 0, stream>>>(wi0, DIN / 32,  wp0);
    pack_w<<<dim3((64 * (HDIM / 32) * 32) / 256), dim3(256), 0, stream>>>(hi0, HDIM / 32, wh0);
    pack_w<<<dim3((64 * (HDIM / 32) * 32) / 256), dim3(256), 0, stream>>>(wi1, HDIM / 32, wp1);
    pack_w<<<dim3((64 * (HDIM / 32) * 32) / 256), dim3(256), 0, stream>>>(hi1, HDIM / 32, wh1);
    pack_x<<<dim3((TSTEPS * 32 * 4 * 32) / 256),  dim3(256), 0, stream>>>(x, xp);
    pack_a<<<dim3((32 * 8 * 32) / 256), dim3(256), 0, stream>>>(h0in,      HDIM, HDIM / 32, h0b[0]);
    pack_a<<<dim3((32 * 8 * 32) / 256), dim3(256), 0, stream>>>(h0in + BH, HDIM, HDIM / 32, h1b[0]);

    // Seed cell state (c0 is [L,B,H] contiguous -> both layers at once).
    hipMemcpyAsync(c0w, c0in, 2 * BH * sizeof(float),
                   hipMemcpyDeviceToDevice, stream);

    dim3 grid(BATCH / MT, HDIM / 16);
    dim3 block(128);

    for (int t = 0; t < TSTEPS; ++t) {
        const int p = t & 1;
        // Layer 0: gates = x_t @ wi0 + h0 @ hi0 + b0
        lstm_step<DIN><<<grid, block, 0, stream>>>(
            xp + (long)t * 32 * 4 * 512, wp0, h0b[p], wh0, b0, c0w, h0b[1 - p]);
        // Layer 1: gates = h0_new @ wi1 + h1 @ hi1 + b1
        lstm_step<HDIM><<<grid, block, 0, stream>>>(
            h0b[1 - p], wp1, h1b[p], wh1, b1, c1w, h1b[1 - p]);
    }

    // T even -> final h1 lives in h1b[0].
    fc_head<<<dim3(BATCH), dim3(256), 0, stream>>>(h1b[0], fcw, fcb, (float*)d_out);
}
